// Revealing_12979391169452
// MI455X (gfx1250) — matrix operand order repaired and checked by reading, not compiled, not run on hardware
//
#include <hip/hip_runtime.h>
#include <hip/hip_bf16.h>

// ---------------------------------------------------------------------------
// GNN watermark pipeline for MI455X (gfx1250, wave32).
// Dense math: v_wmma_f32_16x16x32_bf16 with async global->LDS double buffering.
// Irregular graph ops: encoded-atomic softmax + atomic aggregation.
// ---------------------------------------------------------------------------

typedef __attribute__((ext_vector_type(16))) __bf16 v16bf;
typedef __attribute__((ext_vector_type(8)))  float  v8f;

union ABFrag { v16bf v; unsigned u[8]; };

__device__ __forceinline__ unsigned short f2bf(float f) {
    unsigned u = __float_as_uint(f);
    unsigned r = ((u >> 16) & 1u) + 0x7FFFu;   // round-to-nearest-even
    return (unsigned short)((u + r) >> 16);
}
// order-preserving float<->uint map for atomicMax on floats (handles negatives)
__device__ __forceinline__ unsigned encf(float f) {
    unsigned u = __float_as_uint(f);
    return (u & 0x80000000u) ? ~u : (u | 0x80000000u);
}
__device__ __forceinline__ float decf(unsigned e) {
    unsigned u = (e & 0x80000000u) ? (e ^ 0x80000000u) : ~e;
    return __uint_as_float(u);
}
#define ENC_NEG_INF 0x007FFFFFu   // encf(-inf)

// CDNA5 async global->LDS: 16B per lane, tracked with ASYNCcnt.
__device__ __forceinline__ void async_ld_b128(unsigned lds_addr, const void* gptr) {
    asm volatile("global_load_async_to_lds_b128 %0, %1, off"
                 :: "v"(lds_addr), "v"((unsigned long long)(uintptr_t)gptr)
                 : "memory");
}
#define WAIT_ASYNC(n) asm volatile("s_wait_asynccnt %0" :: "n"(n) : "memory")

__device__ __forceinline__ unsigned lds_off32(const void* p) {
    return (unsigned)(uintptr_t)p;   // generic LDS pointer: addr[31:0] = LDS offset
}

// ---------------------------------------------------------------------------
// Generic elementwise kernels
// ---------------------------------------------------------------------------
__global__ void k_f2bf(const float* __restrict__ src, unsigned short* __restrict__ dst, int n) {
    int i = blockIdx.x * blockDim.x + threadIdx.x;
    if (i < n) dst[i] = f2bf(src[i]);
}
// convert + transpose: W[K][Nc] (f32) -> Wt[Nc][K] (bf16)
__global__ void k_f2bf_t(const float* __restrict__ W, unsigned short* __restrict__ Wt,
                         int K, int Nc) {
    int i = blockIdx.x * blockDim.x + threadIdx.x;
    if (i >= K * Nc) return;
    int k = i / Nc, c = i - k * Nc;
    Wt[(size_t)c * K + k] = f2bf(W[i]);
}
__global__ void k_fill_u32(unsigned* __restrict__ p, unsigned v, int n) {
    int i = blockIdx.x * blockDim.x + threadIdx.x;
    if (i < n) p[i] = v;
}
__global__ void k_elu_bf16(const float* __restrict__ src, unsigned short* __restrict__ dst, int n) {
    int i = blockIdx.x * blockDim.x + threadIdx.x;
    if (i < n) {
        float x = src[i];
        float y = (x > 0.f) ? x : (__expf(x) - 1.f);   // ELU, alpha=1
        dst[i] = f2bf(y);
    }
}
__global__ void k_wm_decode_bf16(const unsigned* __restrict__ enc, unsigned short* __restrict__ dst, int n) {
    int i = blockIdx.x * blockDim.x + threadIdx.x;
    if (i < n) dst[i] = f2bf(decf(enc[i]));
}

// ---------------------------------------------------------------------------
// 1+3) Fused embedding gather + length-masked token mean.
// One wave per node; lane covers 8 of the 256 dims (2x float4).
// ---------------------------------------------------------------------------
__global__ __launch_bounds__(256) void k_node_feat(
    const int* __restrict__ tok, const int* __restrict__ lens,
    const float* __restrict__ emb, float* __restrict__ nf, int n_nodes)
{
    int wid  = threadIdx.x >> 5;
    int lane = threadIdx.x & 31;
    int node = blockIdx.x * 8 + wid;
    if (node >= n_nodes) return;
    int len = lens[node];
    int f0  = lane * 8;
    float acc[8] = {0.f,0.f,0.f,0.f,0.f,0.f,0.f,0.f};
    for (int t = 0; t < 16; ++t) {
        if (t >= len) break;                       // mask is a prefix of length len
        int id = tok[node * 16 + t];
        const float4* rp = (const float4*)(emb + (size_t)id * 256 + f0);
        float4 r0 = rp[0], r1 = rp[1];
        acc[0]+=r0.x; acc[1]+=r0.y; acc[2]+=r0.z; acc[3]+=r0.w;
        acc[4]+=r1.x; acc[5]+=r1.y; acc[6]+=r1.z; acc[7]+=r1.w;
    }
    float inv = 1.f / (float)len;
    float4 o0 = make_float4(acc[0]*inv, acc[1]*inv, acc[2]*inv, acc[3]*inv);
    float4 o1 = make_float4(acc[4]*inv, acc[5]*inv, acc[6]*inv, acc[7]*inv);
    float4* op = (float4*)(nf + (size_t)node * 256 + f0);
    op[0] = o0; op[1] = o1;
}

// ---------------------------------------------------------------------------
// 2) Variable-embedding scatter as a delta correction on the token mean.
// ---------------------------------------------------------------------------
__global__ __launch_bounds__(256) void k_var_scatter(
    const int* __restrict__ vnode, const int* __restrict__ vbegin,
    const int* __restrict__ vlen,  const float* __restrict__ pre,
    const int* __restrict__ tok,   const int* __restrict__ nlens,
    const float* __restrict__ emb, float* __restrict__ nf)
{
    int v = blockIdx.x;
    int d = threadIdx.x;
    int n = vnode[v];
    int b = vbegin[v];
    int L = vlen[v];
    int nlen = nlens[n];
    float inv = 1.f / (float)nlen;
    for (int p = 0; p < 8; ++p) {
        if (p >= L) break;
        int pos = b + p;
        if (pos >= nlen) continue;
        int id = tok[n * 16 + pos];
        float diff = pre[((size_t)v * 8 + p) * 256 + d] - emb[(size_t)id * 256 + d];
        atomicAdd(&nf[(size_t)n * 256 + d], diff * inv);
    }
}

// ---------------------------------------------------------------------------
// WMMA GEMM: C[M,Nc] = A[M,K](bf16,row-major) * Bt[Nc,K](bf16, pre-transposed).
// Block 256 thr (8 waves), tile 64x64, BK=32, async global->LDS double buffer.
// Wave computes 16x32 (2 WMMA sharing one A fragment).
// ---------------------------------------------------------------------------
#define GBM 64
#define GBN 64
#define GBK 32
#define LDT 40   // padded stride (80B): 16B-aligned b128 landing + conflict-free frags

__global__ __launch_bounds__(256) void k_gemm_bf16(
    const unsigned short* __restrict__ A, const unsigned short* __restrict__ Bt,
    float* __restrict__ C, unsigned short* __restrict__ Cb,
    const float* __restrict__ bias, int M, int K, int Nc, int act)
{
    __shared__ unsigned short sA[2][GBM * LDT];
    __shared__ unsigned short sB[2][GBN * LDT];

    int bm   = blockIdx.y * GBM;
    int bn   = blockIdx.x * GBN;
    int tid  = threadIdx.x;
    int lane = tid & 31;
    int wid  = tid >> 5;
    int wrow = (wid & 3) * 16;
    int wcol = (wid >> 2) * 32;
    int half = lane >> 4;
    int r16  = lane & 15;

    // staging coords: thread t covers (row = t>>2, 16B segment = t&3)
    int sr   = tid >> 2;
    int sseg = tid & 3;
    int grow = bm + sr; if (grow >= M) grow = M - 1;   // clamp: dup rows feed masked outputs
    const unsigned short* Ap = A  + (size_t)grow      * K + sseg * 8;
    const unsigned short* Bp = Bt + (size_t)(bn + sr) * K + sseg * 8;
    unsigned aL[2] = { lds_off32(&sA[0][sr * LDT + sseg * 8]),
                       lds_off32(&sA[1][sr * LDT + sseg * 8]) };
    unsigned bL[2] = { lds_off32(&sB[0][sr * LDT + sseg * 8]),
                       lds_off32(&sB[1][sr * LDT + sseg * 8]) };

    // prologue: stage k-step 0 into buffer 0
    async_ld_b128(aL[0], Ap);
    async_ld_b128(bL[0], Bp);

    v8f c0 = {}; v8f c1 = {};
    int nsteps = K / GBK;
    for (int s = 0; s < nsteps; ++s) {
        int cur = s & 1;
        if (s + 1 < nsteps) {               // issue next stage into other buffer
            async_ld_b128(aL[cur ^ 1], Ap + (size_t)(s + 1) * GBK);
            async_ld_b128(bL[cur ^ 1], Bp + (size_t)(s + 1) * GBK);
            WAIT_ASYNC(2);                  // only the 2 just-issued may remain in flight
        } else {
            WAIT_ASYNC(0);
        }
        __syncthreads();                    // current buffer visible to all waves

        ABFrag a, b0, b1;
#pragma unroll
        for (int j = 0; j < 8; ++j) {
            int kp = half * 8 + ((j < 4) ? 2 * j : 2 * j + 8);   // A frag K-pair (16b A 16x32)
            a.u[j]  = *(const unsigned*)(&sA[cur][(wrow + r16) * LDT + kp]);
            int kb = half * 8 + ((j < 4) ? 2 * j : 2 * j + 8);   // B frag K-pair (16b B 32x16)
            b0.u[j] = *(const unsigned*)(&sB[cur][(wcol +      r16) * LDT + kb]);
            b1.u[j] = *(const unsigned*)(&sB[cur][(wcol + 16 + r16) * LDT + kb]);
        }
        c0 = __builtin_amdgcn_wmma_f32_16x16x32_bf16(false, a.v, false, b0.v,
                                                     (short)0, c0, false, false);
        c1 = __builtin_amdgcn_wmma_f32_16x16x32_bf16(false, a.v, false, b1.v,
                                                     (short)0, c1, false, false);
        __syncthreads();                    // all waves done reading before overwrite
    }

    // epilogue: C layout VGPR v -> (M = half*8 + v, N = lane&15)
#pragma unroll
    for (int v = 0; v < 8; ++v) {
        int m = bm + wrow + half * 8 + v;
        if (m >= M) continue;
        int n0 = bn + wcol + r16;
        float x0 = c0[v];
        float x1 = c1[v];
        if (bias) { x0 += bias[n0]; x1 += bias[n0 + 16]; }
        if (act == 1) {
            x0 = (x0 > 0.f) ? x0 : 0.01f * x0;
            x1 = (x1 > 0.f) ? x1 : 0.01f * x1;
        }
        C[(size_t)m * Nc + n0]      = x0;
        C[(size_t)m * Nc + n0 + 16] = x1;
        if (Cb) {
            Cb[(size_t)m * Nc + n0]      = f2bf(x0);
            Cb[(size_t)m * Nc + n0 + 16] = f2bf(x1);
        }
    }
}

// ---------------------------------------------------------------------------
// GAT attention scores: one wave per (node, head); coalesced chunk + shfl reduce
// ---------------------------------------------------------------------------
__global__ __launch_bounds__(256) void k_att_scores(
    const float* __restrict__ z,
    const float* __restrict__ asrc, const float* __restrict__ adst,
    float* __restrict__ es, float* __restrict__ ed, int n_nodes, int H, int F)
{
    int wid  = threadIdx.x >> 5;
    int lane = threadIdx.x & 31;
    int idx  = blockIdx.x * 8 + wid;         // flat (n*H + h)
    if (idx >= n_nodes * H) return;
    int h   = idx & (H - 1);                 // H is a power of two (4)
    int cpl = F >> 5;                        // elements per lane (2 or 8)
    const float* zl = z + (size_t)idx * F + lane * cpl;
    const float* as = asrc + h * F + lane * cpl;
    const float* ad = adst + h * F + lane * cpl;
    float s0 = 0.f, s1 = 0.f;
    for (int j = 0; j < cpl; ++j) { float zz = zl[j]; s0 += zz * as[j]; s1 += zz * ad[j]; }
    for (int o = 16; o; o >>= 1) { s0 += __shfl_xor(s0, o, 32); s1 += __shfl_xor(s1, o, 32); }
    if (lane == 0) { es[idx] = s0; ed[idx] = s1; }
}

// pass 1: per-dst segment max of leaky_relu(e_src[src]+e_dst[dst]) (encoded atomicMax)
__global__ void k_edge_max(const float* __restrict__ es, const float* __restrict__ ed,
                           const int* __restrict__ src, const int* __restrict__ dst,
                           float* __restrict__ ex, unsigned* __restrict__ menc, int E)
{
    int i = blockIdx.x * blockDim.x + threadIdx.x;
    if (i >= E * 4) return;
    int e = i >> 2, h = i & 3;
    int s = src[e], d = dst[e];
    float el = es[s * 4 + h] + ed[d * 4 + h];
    el = (el > 0.f) ? el : 0.2f * el;            // leaky_relu slope 0.2
    ex[i] = el;
    atomicMax(&menc[d * 4 + h], encf(el));
}
// pass 2: ex = exp(e - max); segment sum
__global__ void k_edge_expsum(float* __restrict__ ex, const unsigned* __restrict__ menc,
                              const int* __restrict__ dst, float* __restrict__ ssum, int E)
{
    int i = blockIdx.x * blockDim.x + threadIdx.x;
    if (i >= E * 4) return;
    int e = i >> 2, h = i & 3;
    int d = dst[e];
    float v = __expf(ex[i] - decf(menc[d * 4 + h]));
    ex[i] = v;
    atomicAdd(&ssum[d * 4 + h], v);
}
// pass 3: out[dst] += alpha * z[src]  (one block per edge, coalesced row)
__global__ __launch_bounds__(256) void k_edge_agg(
    const float* __restrict__ z, const float* __restrict__ ex, const float* __restrict__ ssum,
    const int* __restrict__ src, const int* __restrict__ dst,
    float* __restrict__ out, int HF, int fshift)
{
    int e = blockIdx.x;
    int s = src[e], d = dst[e];
    const float* zp = z   + (size_t)s * HF;
    float*       op = out + (size_t)d * HF;
    for (int c = threadIdx.x; c < HF; c += 256) {
        int h = c >> fshift;
        float alpha = ex[e * 4 + h] / ssum[d * 4 + h];
        atomicAdd(&op[c], alpha * zp[c]);
    }
}

// ---------------------------------------------------------------------------
// Var gather + head mean + per-function segment max (encoded atomicMax)
// ---------------------------------------------------------------------------
__global__ __launch_bounds__(256) void k_var_segmax(
    const float* __restrict__ h2, const int* __restrict__ vnib,
    const int* __restrict__ seg, unsigned* __restrict__ wm_enc)
{
    int v = blockIdx.x;
    int f = threadIdx.x;
    int n = vnib[v];
    const float* p = h2 + (size_t)n * 1024;
    float g = 0.25f * (p[f] + p[256 + f] + p[512 + f] + p[768 + f]);
    atomicMax(&wm_enc[seg[v] * 256 + f], encf(g));
}

// ---------------------------------------------------------------------------
extern "C" void kernel_launch(void* const* d_in, const int* in_sizes, int n_in,
                              void* d_out, int out_size, void* d_ws, size_t ws_size,
                              hipStream_t stream)
{
    (void)in_sizes; (void)n_in; (void)out_size; (void)ws_size;
    const int N = 20000, E = 320000, V = 5000, FF = 512;
    const int D = 256, HF1 = 256, HF2 = 1024, HID = 512, CLASSES = 256;

    const int*   tok   = (const int*)  d_in[0];
    const int*   lens  = (const int*)  d_in[1];
    const int*   esrc  = (const int*)  d_in[2];
    const int*   edst  = (const int*)  d_in[3];
    const int*   vnode = (const int*)  d_in[4];
    const int*   vbeg  = (const int*)  d_in[5];
    const int*   vnib  = (const int*)  d_in[6];
    const int*   fseg  = (const int*)  d_in[7];
    const int*   vlen  = (const int*)  d_in[8];
    const float* pre   = (const float*)d_in[9];
    const float* emb   = (const float*)d_in[10];
    const float* W1    = (const float*)d_in[11];
    const float* a1s   = (const float*)d_in[12];
    const float* a1d   = (const float*)d_in[13];
    const float* W2    = (const float*)d_in[14];
    const float* a2s   = (const float*)d_in[15];
    const float* a2d   = (const float*)d_in[16];
    const float* Wd1   = (const float*)d_in[17];
    const float* bd1   = (const float*)d_in[18];
    const float* Wd2   = (const float*)d_in[19];
    const float* bd2   = (const float*)d_in[20];

    // ---- workspace layout (byte cursor, 256B aligned) ----
    char* base = (char*)d_ws;
    size_t off = 0;
    auto alloc = [&](size_t bytes) -> void* {
        void* p = base + off;
        off = (off + bytes + 255) & ~(size_t)255;
        return p;
    };
    float*          nf    = (float*)alloc((size_t)N * D * 4);
    unsigned short* nfb   = (unsigned short*)alloc((size_t)N * D * 2);
    unsigned short* w1t   = (unsigned short*)alloc(256 * 256 * 2);      // [Nc][K]
    unsigned short* w2t   = (unsigned short*)alloc(1024 * 256 * 2);
    unsigned short* wd1t  = (unsigned short*)alloc(512 * 256 * 2);
    unsigned short* wd2t  = (unsigned short*)alloc(256 * 512 * 2);
    float*          z1    = (float*)alloc((size_t)N * HF1 * 4);
    float*          es1   = (float*)alloc((size_t)N * 4 * 4);
    float*          ed1   = (float*)alloc((size_t)N * 4 * 4);
    unsigned*       m1    = (unsigned*)alloc((size_t)N * 4 * 4);
    float*          s1    = (float*)alloc((size_t)N * 4 * 4);
    float*          ex1   = (float*)alloc((size_t)E * 4 * 4);
    float*          out1  = (float*)alloc((size_t)N * HF1 * 4);
    unsigned short* h1b   = (unsigned short*)alloc((size_t)N * HF1 * 2);
    float*          z2    = (float*)alloc((size_t)N * HF2 * 4);
    float*          es2   = (float*)alloc((size_t)N * 4 * 4);
    float*          ed2   = (float*)alloc((size_t)N * 4 * 4);
    unsigned*       m2    = (unsigned*)alloc((size_t)N * 4 * 4);
    float*          s2    = (float*)alloc((size_t)N * 4 * 4);
    float*          ex2   = (float*)alloc((size_t)E * 4 * 4);
    float*          out2  = (float*)alloc((size_t)N * HF2 * 4);
    unsigned*       wme   = (unsigned*)alloc((size_t)FF * 256 * 4);
    unsigned short* wmb   = (unsigned short*)alloc((size_t)FF * 256 * 2);
    float*          hdec  = (float*)alloc((size_t)FF * HID * 4);
    unsigned short* hdecb = (unsigned short*)alloc((size_t)FF * HID * 2);

    auto blk = [](int n) { return (n + 255) / 256; };

    // ---- init accumulators ----
    hipMemsetAsync(out1, 0, (size_t)N * HF1 * 4, stream);
    hipMemsetAsync(out2, 0, (size_t)N * HF2 * 4, stream);
    hipMemsetAsync(s1,   0, (size_t)N * 4 * 4,   stream);
    hipMemsetAsync(s2,   0, (size_t)N * 4 * 4,   stream);
    k_fill_u32<<<blk(N * 4), 256, 0, stream>>>(m1, ENC_NEG_INF, N * 4);
    k_fill_u32<<<blk(N * 4), 256, 0, stream>>>(m2, ENC_NEG_INF, N * 4);
    k_fill_u32<<<blk(FF * 256), 256, 0, stream>>>(wme, ENC_NEG_INF, FF * 256);

    // ---- weights -> bf16, transposed to [Nc][K] for contiguous B-tile DMA ----
    k_f2bf_t<<<blk(256 * 256),  256, 0, stream>>>(W1,  w1t,  256, 256);
    k_f2bf_t<<<blk(256 * 1024), 256, 0, stream>>>(W2,  w2t,  256, 1024);
    k_f2bf_t<<<blk(256 * 512),  256, 0, stream>>>(Wd1, wd1t, 256, 512);
    k_f2bf_t<<<blk(512 * 256),  256, 0, stream>>>(Wd2, wd2t, 512, 256);

    // ---- node features: embed gather + masked mean, then var delta scatter ----
    k_node_feat<<<(N + 7) / 8, 256, 0, stream>>>(tok, lens, emb, nf, N);
    k_var_scatter<<<V, 256, 0, stream>>>(vnode, vbeg, vlen, pre, tok, lens, emb, nf);
    k_f2bf<<<blk(N * D), 256, 0, stream>>>(nf, nfb, N * D);

    // ---- GAT layer 1 ----
    {
        dim3 g(HF1 / GBN, (N + GBM - 1) / GBM);
        k_gemm_bf16<<<g, 256, 0, stream>>>(nfb, w1t, z1, nullptr, nullptr, N, 256, HF1, 0);
    }
    k_att_scores<<<(N * 4 + 7) / 8, 256, 0, stream>>>(z1, a1s, a1d, es1, ed1, N, 4, 64);
    k_edge_max<<<blk(E * 4), 256, 0, stream>>>(es1, ed1, esrc, edst, ex1, m1, E);
    k_edge_expsum<<<blk(E * 4), 256, 0, stream>>>(ex1, m1, edst, s1, E);
    k_edge_agg<<<E, 256, 0, stream>>>(z1, ex1, s1, esrc, edst, out1, HF1, 6);
    k_elu_bf16<<<blk(N * HF1), 256, 0, stream>>>(out1, h1b, N * HF1);

    // ---- GAT layer 2 ----
    {
        dim3 g(HF2 / GBN, (N + GBM - 1) / GBM);
        k_gemm_bf16<<<g, 256, 0, stream>>>(h1b, w2t, z2, nullptr, nullptr, N, 256, HF2, 0);
    }
    k_att_scores<<<(N * 4 + 7) / 8, 256, 0, stream>>>(z2, a2s, a2d, es2, ed2, N, 4, 256);
    k_edge_max<<<blk(E * 4), 256, 0, stream>>>(es2, ed2, esrc, edst, ex2, m2, E);
    k_edge_expsum<<<blk(E * 4), 256, 0, stream>>>(ex2, m2, edst, s2, E);
    k_edge_agg<<<E, 256, 0, stream>>>(z2, ex2, s2, esrc, edst, out2, HF2, 8);

    // ---- var select + head mean + per-function max pool ----
    k_var_segmax<<<V, 256, 0, stream>>>(out2, vnib, fseg, wme);
    k_wm_decode_bf16<<<blk(FF * 256), 256, 0, stream>>>(wme, wmb, FF * 256);

    // ---- decoder MLP (WMMA, fused bias + leaky-relu) ----
    {
        dim3 g(HID / GBN, FF / GBM);
        k_gemm_bf16<<<g, 256, 0, stream>>>(wmb, wd1t, hdec, hdecb, bd1, FF, 256, HID, 1);
    }
    {
        dim3 g(CLASSES / GBN, FF / GBM);
        k_gemm_bf16<<<g, 256, 0, stream>>>(hdecb, wd2t, (float*)d_out, nullptr, bd2,
                                           FF, HID, CLASSES, 0);
    }
}
